// UMamba_512_60327110639686
// MI455X (gfx1250) — compile-verified, matrix operand order checked by reading, not run on hardware
//
#include <hip/hip_runtime.h>
#include <hip/hip_bf16.h>

typedef _Float16 half_t;
typedef __attribute__((ext_vector_type(16))) _Float16 v16h;
typedef __attribute__((ext_vector_type(8)))  _Float16 v8h;
typedef __attribute__((ext_vector_type(8)))  float    v8f;

#define Bsz   8
#define CMID  16
#define DM    256
#define L     4096          // 64*64 tokens
#define DI    512
#define NH    8
#define HD    64
#define NS    128
#define KC    4
#define CONVDIM 768         // DI + 2*NS
#define PROJ  1288          // 2*DI + 2*NS + NH
#define NSPLIT 4
#define NCHUNK (NS / NSPLIT)

// ---------------------------------------------------------------------------
// conv 3x3 stride 2 pad 1:  x[8][3][256][256] -> h[8][16][128][128] (+bias)
// ---------------------------------------------------------------------------
__global__ __launch_bounds__(256) void conv3x3_s2(
    const float* __restrict__ x, const float* __restrict__ w,
    const float* __restrict__ bias, float* __restrict__ out)
{
  int idx = blockIdx.x * 256 + threadIdx.x;          // 2097152
  if (idx >= Bsz * CMID * 128 * 128) return;
  int ow = idx & 127; int t = idx >> 7;
  int oh = t & 127;   t >>= 7;
  int c  = t & 15;    int b = t >> 4;
  float acc = bias[c];
  int ih0 = oh * 2 - 1, iw0 = ow * 2 - 1;
  #pragma unroll
  for (int ci = 0; ci < 3; ++ci)
    #pragma unroll
    for (int kh = 0; kh < 3; ++kh) {
      int ih = ih0 + kh;
      if (ih < 0 || ih >= 256) continue;
      #pragma unroll
      for (int kw = 0; kw < 3; ++kw) {
        int iw = iw0 + kw;
        if (iw < 0 || iw >= 256) continue;
        acc += x[((b * 3 + ci) * 256 + ih) * 256 + iw] *
               w[((c * 3 + ci) * 3 + kh) * 3 + kw];
      }
    }
  out[idx] = acc;
}

// ---------------------------------------------------------------------------
// BatchNorm stats: one WG per channel, deterministic tree reduction
// ---------------------------------------------------------------------------
__global__ __launch_bounds__(256) void bn_stats(
    const float* __restrict__ h, float* __restrict__ stats /*[16][2]*/)
{
  int c = blockIdx.x;
  float s = 0.f, ss = 0.f;
  for (int i = threadIdx.x; i < Bsz * 16384; i += 256) {
    int b = i >> 14; int sp = i & 16383;
    float v = h[((b * CMID + c) << 14) + sp];
    s += v; ss += v * v;
  }
  __shared__ float rs[256], rss[256];
  rs[threadIdx.x] = s; rss[threadIdx.x] = ss;
  __syncthreads();
  for (int off = 128; off; off >>= 1) {
    if (threadIdx.x < off) {
      rs[threadIdx.x]  += rs[threadIdx.x + off];
      rss[threadIdx.x] += rss[threadIdx.x + off];
    }
    __syncthreads();
  }
  if (threadIdx.x == 0) {
    float mean = rs[0] * (1.f / 131072.f);
    float var  = rss[0] * (1.f / 131072.f) - mean * mean;
    stats[c * 2]     = mean;
    stats[c * 2 + 1] = rsqrtf(var + 1e-5f);
  }
}

__global__ __launch_bounds__(256) void bn_gelu(
    float* __restrict__ h, const float* __restrict__ stats,
    const float* __restrict__ gamma, const float* __restrict__ beta)
{
  int idx = blockIdx.x * 256 + threadIdx.x;
  if (idx >= Bsz * CMID * 128 * 128) return;
  int c = (idx >> 14) & 15;
  float v = (h[idx] - stats[c * 2]) * stats[c * 2 + 1] * gamma[c] + beta[c];
  h[idx] = 0.5f * v * (1.f + erff(v * 0.70710678118f));
}

// ---------------------------------------------------------------------------
// im2col for 2x2/s2 patch embed:  A[m][k], m = b*4096 + gh*64 + gw,
// k = c*4 + p*2 + q  (matches patch_w layout (DM, CMID, P, P))
// ---------------------------------------------------------------------------
__global__ __launch_bounds__(256) void im2col_patch(
    const float* __restrict__ h, half_t* __restrict__ Ap)
{
  int idx = blockIdx.x * 256 + threadIdx.x;          // 32768*64
  if (idx >= Bsz * L * 64) return;
  int k = idx & 63; int m = idx >> 6;
  int q = k & 1, p = (k >> 1) & 1, c = k >> 2;
  int gw = m & 63, gh = (m >> 6) & 63, b = m >> 12;
  Ap[idx] = (half_t)h[((b * CMID + c) * 128 + gh * 2 + p) * 128 + gw * 2 + q];
}

// ---------------------------------------------------------------------------
// generic f32 -> f16 convert
// ---------------------------------------------------------------------------
__global__ __launch_bounds__(256) void f32_to_f16(
    const float* __restrict__ src, half_t* __restrict__ dst, int n)
{
  int i = blockIdx.x * 256 + threadIdx.x;
  if (i < n) dst[i] = (half_t)src[i];
}

// ---------------------------------------------------------------------------
// WMMA GEMM:  C[M][N] (+bias,+resid) = A[M][K] * Bw[N][K]^T
// 256 threads = 8 waves; WG tile 32(M) x 128(N); wave tile 16x32
// (one A fragment feeds two B fragments -> 2 v_wmma per K-chunk per wave).
// K staged through LDS in chunks of 32 via global_load_async_to_lds_b128.
// Ragged N handled by clamping staged rows (bad columns never stored).
// Requires M % 32 == 0 and K % 32 == 0 (true for all call sites).
// Epilogue specialized at compile time (no runtime pointer dispatch).
// ---------------------------------------------------------------------------
template <bool HAS_BIAS, bool HAS_RESID, bool OUT_F32, bool OUT_F16>
__global__ __launch_bounds__(256) void gemm_f16(
    const half_t* __restrict__ A, const half_t* __restrict__ Bw,
    const float* __restrict__ bias, const float* __restrict__ resid,
    float* __restrict__ C, half_t* __restrict__ Ch,
    int M, int N, int K)
{
  __shared__ half_t As[32][40];    // pad 40 halves: row*80B keeps 16B align
  __shared__ half_t Bs[128][40];
  int tid  = threadIdx.x;
  int wave = tid >> 5;
  int lane = tid & 31;
  int m0 = blockIdx.y * 32;
  int n0 = blockIdx.x * 128;

  int half16 = lane >> 4;            // which K-half this lane holds
  int l15    = lane & 15;
  int ar  = (wave & 1)  * 16 + l15;  // A row within WG tile
  int brb = (wave >> 1) * 32;        // base B row (2 x 16-wide tiles)

  // staging coordinates (8 halves = 16B per lane-op)
  int sa_r = tid >> 2, sa_c = (tid & 3) * 8;   // A: first 128 lanes

  v8f acc0 = {}, acc1 = {};

  for (int k0 = 0; k0 < K; k0 += 32) {
    if (tid < 128) {
      unsigned lds = (unsigned)(size_t)&As[sa_r][sa_c];
      const half_t* gp = A + (size_t)(m0 + sa_r) * K + k0 + sa_c;
      asm volatile("global_load_async_to_lds_b128 %0, %1, off"
                   :: "v"(lds), "v"(gp) : "memory");
    }
    #pragma unroll
    for (int rep = 0; rep < 2; ++rep) {
      int v = tid + rep * 256;                  // 512 vectors total
      int r = v >> 2, c = (v & 3) * 8;
      int gn = min(n0 + r, N - 1);              // branchless clamp
      unsigned lds = (unsigned)(size_t)&Bs[r][c];
      const half_t* gp = Bw + (size_t)gn * K + k0 + c;
      asm volatile("global_load_async_to_lds_b128 %0, %1, off"
                   :: "v"(lds), "v"(gp) : "memory");
    }
    asm volatile("s_wait_asynccnt 0" ::: "memory");
    __syncthreads();

    // fragments per ISA 7.12.2 (16-bit A 16x32 layout); b128 LDS reads
    v16h av, bv0, bv1;
    #pragma unroll
    for (int j = 0; j < 8; ++j) {
      av[j]      = As[ar][half16 * 8 + j];
      av[j + 8]  = As[ar][16 + half16 * 8 + j];
      bv0[j]     = Bs[brb + l15][half16 * 8 + j];
      bv0[j + 8] = Bs[brb + l15][16 + half16 * 8 + j];
      bv1[j]     = Bs[brb + 16 + l15][half16 * 8 + j];
      bv1[j + 8] = Bs[brb + 16 + l15][16 + half16 * 8 + j];
    }
    acc0 = __builtin_amdgcn_wmma_f32_16x16x32_f16(
        false, av, false, bv0, (short)0, acc0, false, false);
    acc1 = __builtin_amdgcn_wmma_f32_16x16x32_f16(
        false, av, false, bv1, (short)0, acc1, false, false);
    __syncthreads();
  }

  // C/D layout: VGPR v, lanes 0-15 -> M=v, lanes 16-31 -> M=v+8; N = lane&15
  int sm = half16 * 8;
  int mb = m0 + (wave & 1) * 16 + sm;
  #pragma unroll
  for (int tile = 0; tile < 2; ++tile) {
    int n = n0 + brb + tile * 16 + l15;
    if (n >= N) continue;
    float bv = HAS_BIAS ? bias[n] : 0.f;
    #pragma unroll
    for (int v = 0; v < 8; ++v) {
      int m = mb + v;
      if (m >= M) continue;
      float val = (tile ? acc1[v] : acc0[v]) + bv;
      if (HAS_RESID) val += resid[(size_t)m * N + n];
      if (OUT_F32) C[(size_t)m * N + n]  = val;
      if (OUT_F16) Ch[(size_t)m * N + n] = (half_t)val;
    }
  }
}

// ---------------------------------------------------------------------------
// add 2D sincos positional embedding; also emit f16 copy for in_proj GEMM
// ---------------------------------------------------------------------------
__global__ __launch_bounds__(256) void add_pos(
    float* __restrict__ tok, half_t* __restrict__ tokh)
{
  int idx = blockIdx.x * 256 + threadIdx.x;          // 32768*256
  if (idx >= Bsz * L * DM) return;
  int d = idx & 255; int m = idx >> 8; int l = m & (L - 1);
  int r = l >> 6, c = l & 63;
  float pos = (d < 128) ? (float)c : (float)r;
  int dd = d & 127;
  int j  = dd & 63;
  float om = __expf(-(float)j * (9.210340371976184f / 64.f)); // 10000^(-j/64)
  float ang = pos * om;
  float pe = (dd < 64) ? __sinf(ang) : __cosf(ang);
  float v = tok[idx] + pe;
  tok[idx]  = v;
  tokh[idx] = (half_t)v;
}

// ---------------------------------------------------------------------------
// depthwise causal conv1d (K=4, left pad 3) + bias + SiLU over zxbcdt[512:1280]
// ---------------------------------------------------------------------------
__global__ __launch_bounds__(256) void conv1d_silu(
    const float* __restrict__ zxbcdt, const float* __restrict__ w,
    const float* __restrict__ cb, float* __restrict__ xBC)
{
  int idx = blockIdx.x * 256 + threadIdx.x;          // 8*4096*768
  if (idx >= Bsz * L * CONVDIM) return;
  int ch = idx % CONVDIM; int m = idx / CONVDIM;
  int l = m & (L - 1); int b = m >> 12;
  float acc = cb[ch];
  #pragma unroll
  for (int t = 0; t < KC; ++t) {
    int ll = l - (KC - 1) + t;
    if (ll >= 0)
      acc += w[ch * KC + t] *
             zxbcdt[((size_t)(b * L + ll)) * PROJ + DI + ch];
  }
  xBC[idx] = acc / (1.f + __expf(-acc));             // SiLU
}

// ---------------------------------------------------------------------------
// dt = softplus(dt_raw + dt_bias);  dA = exp(-exp(A_log) * dt)
// ---------------------------------------------------------------------------
__global__ __launch_bounds__(256) void dt_kernel(
    const float* __restrict__ zxbcdt, const float* __restrict__ dt_bias,
    const float* __restrict__ A_log, float* __restrict__ dtp,
    float* __restrict__ dAv)
{
  int idx = blockIdx.x * 256 + threadIdx.x;          // 8*4096*8
  if (idx >= Bsz * L * NH) return;
  int h = idx & 7; size_t m = (size_t)(idx >> 3);
  float dt = zxbcdt[m * PROJ + DI + CONVDIM + h] + dt_bias[h];
  float sp = (dt > 20.f) ? dt : log1pf(__expf(dt));
  dtp[idx] = sp;
  dAv[idx] = __expf(-__expf(A_log[h]) * sp);
}

// ---------------------------------------------------------------------------
// SSM scan. Grid = B*NH*NSPLIT = 256 WGs. Each WG owns state s[64][32] for
// one (b,h,n-split); 256 threads keep 8 state elems each in registers;
// partial y reduced per-quad via wave32 shfl_xor.
// ---------------------------------------------------------------------------
__global__ __launch_bounds__(256) void ssm_scan(
    const float* __restrict__ xBC, const float* __restrict__ dtp,
    const float* __restrict__ dAv, float* __restrict__ ypart)
{
  int wg = blockIdx.x;
  int s  = wg & 3;
  int h  = (wg >> 2) & 7;
  int b  = wg >> 5;
  int tid = threadIdx.x;
  int p  = tid >> 2;            // head-dim row 0..63
  int nq = tid & 3;             // quad: 8 n's each

  float st[8];
  #pragma unroll
  for (int j = 0; j < 8; ++j) st[j] = 0.f;

  __shared__ float sh_xh[HD];
  __shared__ float sh_B[NCHUNK];
  __shared__ float sh_C[NCHUNK];
  __shared__ float sh_sc[2];

  float* yp = ypart + (size_t)s * ((size_t)Bsz * L * NH * HD);

  for (int l = 0; l < L; ++l) {
    size_t row = (size_t)b * L + l;
    const float* xrow = xBC + row * CONVDIM;
    if (tid < 64)        sh_xh[tid]      = xrow[h * HD + tid];
    else if (tid < 96)   sh_B[tid - 64]  = xrow[DI + s * NCHUNK + (tid - 64)];
    else if (tid < 128)  sh_C[tid - 96]  = xrow[DI + NS + s * NCHUNK + (tid - 96)];
    else if (tid == 128) sh_sc[0] = dtp[row * NH + h];
    else if (tid == 129) sh_sc[1] = dAv[row * NH + h];
    __syncthreads();

    float da  = sh_sc[1];
    float dtx = sh_sc[0] * sh_xh[p];
    float part = 0.f;
    #pragma unroll
    for (int j = 0; j < 8; ++j) {
      int n = nq * 8 + j;
      float sv = st[j] * da + dtx * sh_B[n];
      st[j] = sv;
      part += sv * sh_C[n];
    }
    part += __shfl_xor(part, 1, 32);
    part += __shfl_xor(part, 2, 32);
    if (nq == 0) yp[(row * NH + h) * HD + p] = part;
    __syncthreads();
  }
}

// ---------------------------------------------------------------------------
// reduce n-splits + D*xh, gate with silu(z), RMSNorm, emit f16 for out_proj
// ---------------------------------------------------------------------------
__global__ __launch_bounds__(256) void gate_rms(
    const float* __restrict__ ypart, const float* __restrict__ xBC,
    const float* __restrict__ zxbcdt, const float* __restrict__ Dp,
    const float* __restrict__ norm_w, half_t* __restrict__ ynh)
{
  size_t row = blockIdx.x;
  int tid = threadIdx.x;
  __shared__ float red[256];
  float vals[2];
  float ssq = 0.f;
  #pragma unroll
  for (int e = 0; e < 2; ++e) {
    int d = tid * 2 + e;                 // 0..511
    int h = d >> 6, p = d & 63;
    float y = Dp[h] * xBC[row * CONVDIM + h * HD + p];
    size_t yi = (row * NH + h) * HD + p;
    #pragma unroll
    for (int s = 0; s < NSPLIT; ++s)
      y += ypart[(size_t)s * ((size_t)Bsz * L * NH * HD) + yi];
    float z = zxbcdt[row * PROJ + d];
    y *= z / (1.f + __expf(-z));         // * silu(z)
    vals[e] = y;
    ssq += y * y;
  }
  red[tid] = ssq;
  __syncthreads();
  for (int off = 128; off; off >>= 1) {
    if (tid < off) red[tid] += red[tid + off];
    __syncthreads();
  }
  float rinv = rsqrtf(red[0] * (1.f / (float)DI) + 1e-5f);
  #pragma unroll
  for (int e = 0; e < 2; ++e) {
    int d = tid * 2 + e;
    ynh[row * DI + d] = (half_t)(vals[e] * rinv * norm_w[d]);
  }
}

// ---------------------------------------------------------------------------
// unpatchify: o[b*4096+gh*64+gw][(p*2+q)*16+c] -> out[b][c][gh*2+p][gw*2+q]
// ---------------------------------------------------------------------------
__global__ __launch_bounds__(256) void unpatchify(
    const float* __restrict__ o, float* __restrict__ out)
{
  int idx = blockIdx.x * 256 + threadIdx.x;          // 2097152
  if (idx >= Bsz * CMID * 128 * 128) return;
  int ww = idx & 127; int t = idx >> 7;
  int hh = t & 127;   t >>= 7;
  int c  = t & 15;    int b = t >> 4;
  int gh = hh >> 1, p = hh & 1, gw = ww >> 1, q = ww & 1;
  size_t m = (size_t)b * L + gh * 64 + gw;
  out[idx] = o[m * 64 + (p * 2 + q) * CMID + c];
}

// ---------------------------------------------------------------------------
extern "C" void kernel_launch(void* const* d_in, const int* in_sizes, int n_in,
                              void* d_out, int out_size, void* d_ws, size_t ws_size,
                              hipStream_t stream)
{
  (void)in_sizes; (void)n_in; (void)out_size; (void)ws_size;
  const float* x         = (const float*)d_in[0];
  const float* conv_w    = (const float*)d_in[1];
  const float* conv_b    = (const float*)d_in[2];
  const float* bn_gamma  = (const float*)d_in[3];
  const float* bn_beta   = (const float*)d_in[4];
  const float* patch_w   = (const float*)d_in[5];
  const float* patch_b   = (const float*)d_in[6];
  const float* in_proj_w = (const float*)d_in[7];
  const float* conv1d_w  = (const float*)d_in[8];
  const float* conv1d_b  = (const float*)d_in[9];
  const float* dt_bias   = (const float*)d_in[10];
  const float* A_log     = (const float*)d_in[11];
  const float* Dp        = (const float*)d_in[12];
  const float* norm_w    = (const float*)d_in[13];
  const float* out_proj_w= (const float*)d_in[14];
  const float* final_w   = (const float*)d_in[15];
  const float* final_b   = (const float*)d_in[16];
  float* outp = (float*)d_out;

  char*  ws  = (char*)d_ws;
  size_t off = 0;
  auto alloc = [&](size_t bytes) -> void* {
    void* p = ws + off;
    off = (off + bytes + 255) & ~(size_t)255;
    return p;
  };

  const size_t M = (size_t)Bsz * L;                  // 32768 token rows
  float*  h1     = (float*) alloc((size_t)Bsz*CMID*128*128*4);
  float*  stats  = (float*) alloc(256);
  half_t* Ap     = (half_t*)alloc(M * 64 * 2);
  half_t* pw_h   = (half_t*)alloc((size_t)DM*64*2);
  float*  tok    = (float*) alloc(M * DM * 4);
  half_t* tok_h  = (half_t*)alloc(M * DM * 2);
  half_t* ipw_h  = (half_t*)alloc((size_t)PROJ*DM*2);
  float*  zxb    = (float*) alloc(M * PROJ * 4);
  float*  xBC    = (float*) alloc(M * CONVDIM * 4);
  float*  dtp    = (float*) alloc(M * NH * 4);
  float*  dAv    = (float*) alloc(M * NH * 4);
  float*  ypart  = (float*) alloc((size_t)NSPLIT * M * NH * HD * 4);
  half_t* yn_h   = (half_t*)alloc(M * DI * 2);
  half_t* opw_h  = (half_t*)alloc((size_t)DM*DI*2);
  half_t* y2_h   = (half_t*)alloc(M * DM * 2);
  half_t* fw_h   = (half_t*)alloc((size_t)64*DM*2);
  float*  o_buf  = (float*) alloc(M * 64 * 4);

  // 1. stem conv + BN + GELU
  conv3x3_s2<<<8192, 256, 0, stream>>>(x, conv_w, conv_b, h1);
  bn_stats  <<<CMID, 256, 0, stream>>>(h1, stats);
  bn_gelu   <<<8192, 256, 0, stream>>>(h1, stats, bn_gamma, bn_beta);

  // 2. patch embed GEMM (M=32768, N=256, K=64) + pos emb
  im2col_patch<<<8192, 256, 0, stream>>>(h1, Ap);
  f32_to_f16<<<64, 256, 0, stream>>>(patch_w, pw_h, DM * 64);
  gemm_f16<true, false, true, false><<<dim3(2, 1024), 256, 0, stream>>>(
      Ap, pw_h, patch_b, nullptr, tok, nullptr, (int)M, DM, 64);
  add_pos<<<32768, 256, 0, stream>>>(tok, tok_h);

  // 3. in_proj GEMM (M=32768, N=1288, K=256)
  f32_to_f16<<<1288, 256, 0, stream>>>(in_proj_w, ipw_h, PROJ * DM);
  gemm_f16<false, false, true, false><<<dim3(11, 1024), 256, 0, stream>>>(
      tok_h, ipw_h, nullptr, nullptr, zxb, nullptr, (int)M, PROJ, DM);

  // 4. causal depthwise conv1d + SiLU; dt / dA precompute
  conv1d_silu<<<98304, 256, 0, stream>>>(zxb, conv1d_w, conv1d_b, xBC);
  dt_kernel<<<1024, 256, 0, stream>>>(zxb, dt_bias, A_log, dtp, dAv);

  // 5. sequential SSM scan, NS split 4 ways (256 WGs)
  ssm_scan<<<Bsz * NH * NSPLIT, 256, 0, stream>>>(xBC, dtp, dAv, ypart);

  // 6. combine splits + D*xh, gate with silu(z), RMSNorm
  gate_rms<<<(int)M, 256, 0, stream>>>(ypart, xBC, zxb, Dp, norm_w, yn_h);

  // 7. out_proj GEMM (M=32768, N=256, K=512) + tok residual -> f16
  f32_to_f16<<<512, 256, 0, stream>>>(out_proj_w, opw_h, DM * DI);
  gemm_f16<false, true, false, true><<<dim3(2, 1024), 256, 0, stream>>>(
      yn_h, opw_h, nullptr, tok, nullptr, y2_h, (int)M, DM, DI);

  // 8. final GEMM (M=32768, N=64, K=256) + bias
  f32_to_f16<<<64, 256, 0, stream>>>(final_w, fw_h, 64 * DM);
  gemm_f16<true, false, true, false><<<dim3(1, 1024), 256, 0, stream>>>(
      y2_h, fw_h, final_b, nullptr, o_buf, nullptr, (int)M, 64, DM);

  // 9. unpatchify to output layout
  unpatchify<<<8192, 256, 0, stream>>>(o_buf, outp);
}
